// SpellerModel_71090298683500
// MI455X (gfx1250) — compile-verified, matrix operand order checked by reading, not run on hardware
//
#include <hip/hip_runtime.h>
#include <hip/hip_bf16.h>
#include <stdint.h>

typedef __attribute__((ext_vector_type(16))) __bf16 v16bf;
typedef __attribute__((ext_vector_type(8)))  float  v8f;

#define Tn 300
#define Bn 64
#define Un 1024
#define Cn 128
#define Hn 512
#define En 256
#define Vn 33
#define G4H 2048   // 4*H
#define XLD 384    // E + C

union TileU { uint4 u[2]; v16bf v; };

__device__ __forceinline__ v16bf load_tile16(const __bf16* p) {
  // p points at (row base + kblock + 8*half); loads elements [0..7] and [16..23]
  TileU t;
  t.u[0] = *reinterpret_cast<const uint4*>(p);
  t.u[1] = *reinterpret_cast<const uint4*>(p + 16);
  return t.v;
}

__device__ __forceinline__ float sigm(float x) { return 1.0f / (1.0f + __expf(-x)); }

// ---------------------------------------------------------------------------
// Fused LSTM layer: gates = A1@W1^T + A2@W2^T (+bias), then pointwise c/h.
// W1/W2 are bf16, gate-interleaved rows (r_new = 4*j + gate).
// Grid: 16 blocks x 256 threads; block handles 128 gate-columns, M = 64.
// ---------------------------------------------------------------------------
__global__ __launch_bounds__(256) void lstm_layer_kernel(
    const __bf16* __restrict__ A1, int ldA1, int K1,
    const __bf16* __restrict__ W1,          // [2048][K1]
    const __bf16* __restrict__ A2,          // h_prev [64][512]
    const __bf16* __restrict__ W2,          // [2048][512]
    const float*  __restrict__ bias,        // [2048] gate-interleaved (b_ih + b_hh)
    float*        __restrict__ c_state,     // [64][512] in-place
    __bf16*       __restrict__ h_out)       // [64][512] (parity buffer != A2)
{
  __shared__ float gl[64][132];             // gates staging (padded)
  const int wave = threadIdx.x >> 5;
  const int lane = threadIdx.x & 31;
  const int half = lane >> 4;
  const int lm   = lane & 15;
  const int n0   = blockIdx.x * 128;
  const int ncol = n0 + wave * 16 + lm;

  // Warm WGP$ with the recurrent-weight row while phase 1 executes.
  __builtin_prefetch(W2 + (size_t)ncol * Hn, 0, 1);

  v8f acc[4];
  #pragma unroll
  for (int mt = 0; mt < 4; ++mt)
    #pragma unroll
    for (int e = 0; e < 8; ++e) acc[mt][e] = 0.0f;

  // Phase 1: A1 @ W1^T
  const __bf16* wrow = W1 + (size_t)ncol * K1;
  for (int kb = 0; kb < K1; kb += 32) {
    v16bf bt = load_tile16(wrow + kb + 8 * half);
    #pragma unroll
    for (int mt = 0; mt < 4; ++mt) {
      v16bf at = load_tile16(A1 + (size_t)(mt * 16 + lm) * ldA1 + kb + 8 * half);
      acc[mt] = __builtin_amdgcn_wmma_f32_16x16x32_bf16(
          false, at, false, bt, (short)0, acc[mt], false, false);
    }
  }
  // Phase 2: A2 @ W2^T  (recurrent term, K = 512)
  wrow = W2 + (size_t)ncol * Hn;
  for (int kb = 0; kb < Hn; kb += 32) {
    v16bf bt = load_tile16(wrow + kb + 8 * half);
    #pragma unroll
    for (int mt = 0; mt < 4; ++mt) {
      v16bf at = load_tile16(A2 + (size_t)(mt * 16 + lm) * Hn + kb + 8 * half);
      acc[mt] = __builtin_amdgcn_wmma_f32_16x16x32_bf16(
          false, at, false, bt, (short)0, acc[mt], false, false);
    }
  }

  // D layout: lane holds column n = wave*16+lm; VGPR v -> row m = v + 8*half (+16*mt)
  #pragma unroll
  for (int mt = 0; mt < 4; ++mt)
    #pragma unroll
    for (int v = 0; v < 8; ++v)
      gl[mt * 16 + v + 8 * half][wave * 16 + lm] = acc[mt][v];
  __syncthreads();

  // Fused LSTM pointwise: block owns h-indices j in [n0/4, n0/4 + 32)
  for (int p = threadIdx.x; p < 64 * 32; p += 256) {
    int b  = p >> 5;
    int jl = p & 31;
    int col = jl * 4;
    int n   = n0 + col;
    float gi = gl[b][col + 0] + bias[n + 0];
    float gf = gl[b][col + 1] + bias[n + 1];
    float gg = gl[b][col + 2] + bias[n + 2];
    float go = gl[b][col + 3] + bias[n + 3];
    int j = (n0 >> 2) + jl;
    float cc = c_state[b * Hn + j];
    float c2 = sigm(gf) * cc + sigm(gi) * tanhf(gg);
    c_state[b * Hn + j] = c2;
    h_out[b * Hn + j] = (__bf16)(sigm(go) * tanhf(c2));
  }
}

// ---------------------------------------------------------------------------
// Merged attention + loss kernel, 65 blocks x 256 threads:
//   blocks 0..63 : per-batch attention. Each block computes its own q row
//                  (q[b] = h3[b]@Wq^T + bq, 128x512 VALU dot), then
//                  energy = q.key, softmax * mask, L1-normalize,
//                  ctx = attn.val written into X[t+1][:,256:384] (bf16).
//   block 64     : logits = h3@Wv^T + bv via WMMA (N padded 33->48),
//                  -log_softmax(logits)[target] accumulated into lossacc.
// ---------------------------------------------------------------------------
__global__ __launch_bounds__(256) void attn_loss_kernel(
    const __bf16* __restrict__ h3,       // [64][512] bf16
    const __bf16* __restrict__ Wq,       // [128][512] bf16
    const float*  __restrict__ bq,       // [128]
    const __bf16* __restrict__ Wv,       // [48][512] bf16 (rows 33..47 zero)
    const float*  __restrict__ bv,       // [33]
    const float*  __restrict__ key,      // [64][1024][128]
    const float*  __restrict__ val,      // [64][1024][128]
    const float*  __restrict__ amask,    // [64][1024]
    __bf16*       __restrict__ Xnext,    // X + (t+1)*64*384 + 256 (ctx slot)
    int write_ctx,
    const int*    __restrict__ target,   // [T][B]
    const float*  __restrict__ tmask,    // [T][B]
    float*        __restrict__ lossacc,  // [64]
    int t)
{
  __shared__ float qs[Cn];
  __shared__ float hrow[Hn];
  __shared__ float w[Un];
  __shared__ float red[256];
  __shared__ float lg[64][64];
  const int tid = threadIdx.x;

  if (blockIdx.x == Bn) {
    // ---------------- logits + loss block (WMMA) ----------------
    const int wave = tid >> 5;
    const int lane = tid & 31;
    const int half = lane >> 4;
    const int lm   = lane & 15;
    if (wave < 3) {
      v8f acc[4];
      #pragma unroll
      for (int mt = 0; mt < 4; ++mt)
        #pragma unroll
        for (int e = 0; e < 8; ++e) acc[mt][e] = 0.0f;
      const int n = wave * 16 + lm;
      const __bf16* wrow = Wv + (size_t)n * Hn;
      for (int kb = 0; kb < Hn; kb += 32) {
        v16bf bt = load_tile16(wrow + kb + 8 * half);
        #pragma unroll
        for (int mt = 0; mt < 4; ++mt) {
          v16bf at = load_tile16(h3 + (size_t)(mt * 16 + lm) * Hn + kb + 8 * half);
          acc[mt] = __builtin_amdgcn_wmma_f32_16x16x32_bf16(
              false, at, false, bt, (short)0, acc[mt], false, false);
        }
      }
      float bva = (n < Vn) ? bv[n] : 0.0f;
      #pragma unroll
      for (int mt = 0; mt < 4; ++mt)
        #pragma unroll
        for (int v = 0; v < 8; ++v)
          lg[mt * 16 + v + 8 * half][n] = acc[mt][v] + bva;
    }
    __syncthreads();
    if (tid < Bn) {
      int b = tid;
      float mx = lg[b][0];
      for (int n = 1; n < Vn; ++n) mx = fmaxf(mx, lg[b][n]);
      float s = 0.0f;
      for (int n = 0; n < Vn; ++n) s += __expf(lg[b][n] - mx);
      int tg = target[t * Bn + b];
      float lsm = lg[b][tg] - mx - __logf(s);
      lossacc[b] += tmask[t * Bn + b] * (-lsm);
    }
    return;
  }

  // ---------------- attention block for batch row b ----------------
  const int b = blockIdx.x;

  // stage h3[b] row as f32 in LDS
  hrow[tid]       = (float)h3[(size_t)b * Hn + tid];
  hrow[tid + 256] = (float)h3[(size_t)b * Hn + tid + 256];
  __syncthreads();

  // q[b][c] = bq[c] + sum_j hrow[j] * Wq[c][j]
  if (tid < Cn) {
    const __bf16* wr = Wq + (size_t)tid * Hn;
    float a = bq[tid];
    for (int j = 0; j < Hn; ++j) a += hrow[j] * (float)wr[j];
    qs[tid] = a;
  }
  __syncthreads();

  float e[4];
  float lmax = -3.4e38f;
  #pragma unroll
  for (int i = 0; i < 4; ++i) {
    int u = tid + i * 256;
    const float* kr = key + ((size_t)b * Un + u) * Cn;
    float d = 0.0f;
    for (int c = 0; c < Cn; ++c) d += qs[c] * kr[c];
    e[i] = d;
    lmax = fmaxf(lmax, d);
  }
  red[tid] = lmax;
  __syncthreads();
  for (int s = 128; s > 0; s >>= 1) {
    if (tid < s) red[tid] = fmaxf(red[tid], red[tid + s]);
    __syncthreads();
  }
  float mx = red[0];
  __syncthreads();

  float ls1 = 0.0f, ls2 = 0.0f;
  #pragma unroll
  for (int i = 0; i < 4; ++i) {
    int u = tid + i * 256;
    float ex = __expf(e[i] - mx);
    float wm = amask[(size_t)b * Un + u] * ex;
    w[u] = wm;
    ls1 += ex;
    ls2 += wm;
  }
  red[tid] = ls1;
  __syncthreads();
  for (int s = 128; s > 0; s >>= 1) {
    if (tid < s) red[tid] += red[tid + s];
    __syncthreads();
  }
  float s1 = red[0];
  __syncthreads();
  red[tid] = ls2;
  __syncthreads();
  for (int s = 128; s > 0; s >>= 1) {
    if (tid < s) red[tid] += red[tid + s];
    __syncthreads();
  }
  float s2 = red[0];
  __syncthreads();

  // attn_final = mask*exp / (s1 * max(s2/s1, 1e-12))
  float scale = 1.0f / (s1 * fmaxf(s2 / s1, 1e-12f));

  if (write_ctx) {
    int c  = tid & 127;
    int uh = tid >> 7;
    float a = 0.0f;
    for (int u = uh * 512; u < uh * 512 + 512; ++u)
      a += w[u] * val[((size_t)b * Un + u) * Cn + c];
    red[tid] = a * scale;
    __syncthreads();
    if (uh == 0)
      Xnext[(size_t)b * XLD + c] = (__bf16)(red[c] + red[c + 128]);
  }
}

// ---------------------------------------------------------------------------
// Prep / finalize kernels
// ---------------------------------------------------------------------------
__global__ void cvt_weight_kernel(const float* __restrict__ src,
                                  __bf16* __restrict__ dst,
                                  int Ndst, int Nsrc, int K, int reorder) {
  int idx = blockIdx.x * blockDim.x + threadIdx.x;
  if (idx >= Ndst * K) return;
  int r = idx / K, k = idx - r * K;
  int ro = reorder ? ((r & 3) * (Nsrc >> 2) + (r >> 2)) : r;
  float v = (ro < Nsrc) ? src[(size_t)ro * K + k] : 0.0f;
  dst[idx] = (__bf16)v;
}

__global__ void prep_bias_kernel(const float* __restrict__ bih,
                                 const float* __restrict__ bhh,
                                 float* __restrict__ dst, int N) {
  int r = blockIdx.x * blockDim.x + threadIdx.x;
  if (r >= N) return;
  int ro = (r & 3) * (N >> 2) + (r >> 2);
  dst[r] = bih[ro] + bhh[ro];
}

__global__ void prep_embed_kernel(const int* __restrict__ target,
                                  const float* __restrict__ emb,
                                  __bf16* __restrict__ X) {
  int p = blockIdx.x * blockDim.x + threadIdx.x;  // over T*B
  if (p >= Tn * Bn) return;
  int t = p / Bn, b = p - t * Bn;
  int y = (t == 0) ? 0 : target[(t - 1) * Bn + b];
  const float* er = emb + (size_t)y * En;
  __bf16* xr = X + ((size_t)t * Bn + b) * XLD;
  for (int e2 = 0; e2 < En; ++e2) xr[e2] = (__bf16)er[e2];
}

__global__ void prep_ctx0_kernel(const float* __restrict__ hx1,
                                 const float* __restrict__ Wq,
                                 const float* __restrict__ bq,
                                 __bf16* __restrict__ X) {
  int c = threadIdx.x;  // 128 threads
  if (c >= Cn) return;
  float v = bq[c];
  for (int j = 0; j < Hn; ++j) v += hx1[j] * Wq[(size_t)c * Hn + j];
  __bf16 bv16 = (__bf16)v;
  for (int b = 0; b < Bn; ++b) X[(size_t)b * XLD + En + c] = bv16;
}

__global__ void prep_state_kernel(const float* hx1, const float* cx1,
                                  const float* hx2, const float* cx2,
                                  const float* hx3, const float* cx3,
                                  __bf16* __restrict__ hbase,  // [3][2][64][512]
                                  float* __restrict__ cbase,   // [3][64][512]
                                  float* __restrict__ lossacc) {
  int idx = blockIdx.x * blockDim.x + threadIdx.x;  // over 64*512
  if (idx >= Bn * Hn) return;
  int j = idx & (Hn - 1);
  const float* hx[3] = {hx1, hx2, hx3};
  const float* cx[3] = {cx1, cx2, cx3};
  for (int l = 0; l < 3; ++l) {
    hbase[(size_t)(l * 2 + 0) * Bn * Hn + idx] = (__bf16)hx[l][j];  // parity 0
    cbase[(size_t)l * Bn * Hn + idx] = cx[l][j];
  }
  if (idx < Bn) lossacc[idx] = 0.0f;
}

__global__ void finalize_kernel(const float* __restrict__ lossacc,
                                float* __restrict__ out) {
  __shared__ float s[Bn];
  s[threadIdx.x] = lossacc[threadIdx.x];
  __syncthreads();
  for (int k = 32; k > 0; k >>= 1) {
    if (threadIdx.x < k) s[threadIdx.x] += s[threadIdx.x + k];
    __syncthreads();
  }
  if (threadIdx.x == 0) out[0] = s[0] / (float)Bn;
}

// ---------------------------------------------------------------------------
extern "C" void kernel_launch(void* const* d_in, const int* in_sizes, int n_in,
                              void* d_out, int out_size, void* d_ws, size_t ws_size,
                              hipStream_t stream) {
  (void)in_sizes; (void)n_in; (void)out_size; (void)ws_size;

  const int*   target   = (const int*)d_in[0];
  const float* tmask    = (const float*)d_in[1];
  const float* att_key  = (const float*)d_in[2];
  const float* att_val  = (const float*)d_in[3];
  const float* att_mask = (const float*)d_in[4];
  const float* embed    = (const float*)d_in[5];
  const float* W_ih1 = (const float*)d_in[6];  const float* b_ih1 = (const float*)d_in[7];
  const float* W_hh1 = (const float*)d_in[8];  const float* b_hh1 = (const float*)d_in[9];
  const float* W_ih2 = (const float*)d_in[10]; const float* b_ih2 = (const float*)d_in[11];
  const float* W_hh2 = (const float*)d_in[12]; const float* b_hh2 = (const float*)d_in[13];
  const float* W_ih3 = (const float*)d_in[14]; const float* b_ih3 = (const float*)d_in[15];
  const float* W_hh3 = (const float*)d_in[16]; const float* b_hh3 = (const float*)d_in[17];
  const float* Wq = (const float*)d_in[18];    const float* bq = (const float*)d_in[19];
  const float* Wv = (const float*)d_in[20];    const float* bv = (const float*)d_in[21];
  const float* hx1 = (const float*)d_in[22];   const float* cx1 = (const float*)d_in[23];
  const float* hx2 = (const float*)d_in[24];   const float* cx2 = (const float*)d_in[25];
  const float* hx3 = (const float*)d_in[26];   const float* cx3 = (const float*)d_in[27];

  // ---- workspace layout (all chunk sizes are multiples of 256 B) ----
  char* ws = (char*)d_ws;
  size_t off = 0;
  auto take = [&](size_t bytes) { char* p = ws + off; off += bytes; return p; };
  __bf16* W1i = (__bf16*)take((size_t)G4H * XLD * 2);   // 2048x384
  __bf16* W1h = (__bf16*)take((size_t)G4H * Hn * 2);
  __bf16* W2i = (__bf16*)take((size_t)G4H * Hn * 2);
  __bf16* W2h = (__bf16*)take((size_t)G4H * Hn * 2);
  __bf16* W3i = (__bf16*)take((size_t)G4H * Hn * 2);
  __bf16* W3h = (__bf16*)take((size_t)G4H * Hn * 2);
  __bf16* Wqb = (__bf16*)take((size_t)Cn * Hn * 2);     // 128x512
  __bf16* Wvb = (__bf16*)take((size_t)48 * Hn * 2);     // 48x512 (padded)
  float* bias1 = (float*)take(G4H * 4);
  float* bias2 = (float*)take(G4H * 4);
  float* bias3 = (float*)take(G4H * 4);
  __bf16* X     = (__bf16*)take((size_t)Tn * Bn * XLD * 2);   // [T][64][384]
  __bf16* hbase = (__bf16*)take((size_t)3 * 2 * Bn * Hn * 2); // [l][parity][64][512]
  float* cbase   = (float*)take((size_t)3 * Bn * Hn * 4);
  float* lossacc = (float*)take(256);

  auto hb = [&](int l, int p) { return hbase + (size_t)(l * 2 + p) * Bn * Hn; };

  // ---- prep: weight conversion + gate interleave, embeddings, state ----
  const int TPB = 256;
  auto cvt = [&](const float* s, __bf16* d, int Nd, int Ns, int K, int ro) {
    int n = Nd * K;
    cvt_weight_kernel<<<(n + TPB - 1) / TPB, TPB, 0, stream>>>(s, d, Nd, Ns, K, ro);
  };
  cvt(W_ih1, W1i, G4H, G4H, XLD, 1);
  cvt(W_hh1, W1h, G4H, G4H, Hn, 1);
  cvt(W_ih2, W2i, G4H, G4H, Hn, 1);
  cvt(W_hh2, W2h, G4H, G4H, Hn, 1);
  cvt(W_ih3, W3i, G4H, G4H, Hn, 1);
  cvt(W_hh3, W3h, G4H, G4H, Hn, 1);
  cvt(Wq, Wqb, Cn, Cn, Hn, 0);
  cvt(Wv, Wvb, 48, Vn, Hn, 0);
  prep_bias_kernel<<<(G4H + TPB - 1) / TPB, TPB, 0, stream>>>(b_ih1, b_hh1, bias1, G4H);
  prep_bias_kernel<<<(G4H + TPB - 1) / TPB, TPB, 0, stream>>>(b_ih2, b_hh2, bias2, G4H);
  prep_bias_kernel<<<(G4H + TPB - 1) / TPB, TPB, 0, stream>>>(b_ih3, b_hh3, bias3, G4H);
  prep_embed_kernel<<<(Tn * Bn + TPB - 1) / TPB, TPB, 0, stream>>>(target, embed, X);
  prep_ctx0_kernel<<<1, 128, 0, stream>>>(hx1, Wq, bq, X);
  prep_state_kernel<<<(Bn * Hn + TPB - 1) / TPB, TPB, 0, stream>>>(
      hx1, cx1, hx2, cx2, hx3, cx3, hbase, cbase, lossacc);

  // ---- time loop: 4 stream-ordered kernels per step ----
  for (int t = 0; t < Tn; ++t) {
    int pi = t & 1, po = (t + 1) & 1;
    const __bf16* Xt = X + (size_t)t * Bn * XLD;

    lstm_layer_kernel<<<16, 256, 0, stream>>>(
        Xt, XLD, XLD, W1i, hb(0, pi), W1h, bias1, cbase + 0 * Bn * Hn, hb(0, po));
    lstm_layer_kernel<<<16, 256, 0, stream>>>(
        hb(0, po), Hn, Hn, W2i, hb(1, pi), W2h, bias2, cbase + 1 * (size_t)Bn * Hn, hb(1, po));
    lstm_layer_kernel<<<16, 256, 0, stream>>>(
        hb(1, po), Hn, Hn, W3i, hb(2, pi), W3h, bias3, cbase + 2 * (size_t)Bn * Hn, hb(2, po));

    int wc = (t + 1 < Tn) ? 1 : 0;
    __bf16* Xnext = X + ((size_t)((t + 1 < Tn) ? (t + 1) : t) * Bn) * XLD + En;
    attn_loss_kernel<<<Bn + 1, 256, 0, stream>>>(
        hb(2, po), Wqb, bq, Wvb, bv, att_key, att_val, att_mask,
        Xnext, wc, target, tmask, lossacc, t);
  }

  finalize_kernel<<<1, Bn, 0, stream>>>(lossacc, (float*)d_out);
}
